// QLinear_56453050139062
// MI455X (gfx1250) — 8-bit matrix operand order repaired and checked by reading, compiled, not run on hardware
//
#include <hip/hip_runtime.h>


typedef __attribute__((ext_vector_type(16))) int   v16i;
typedef __attribute__((ext_vector_type(8)))  float v8f;
typedef __attribute__((ext_vector_type(4)))  int   v4i;

#define BM 128
#define BN 128
#define BK 128
#define GEMM_THREADS 256

#if __has_builtin(__builtin_amdgcn_global_load_async_to_lds_b128)
#define USE_ASYNC_LDS 1
typedef __attribute__((address_space(1))) v4i* gptr_b128;
typedef __attribute__((address_space(3))) v4i* lptr_b128;
#else
#define USE_ASYNC_LDS 0
#endif

__device__ __forceinline__ void wait_asynccnt0() {
#if __has_builtin(__builtin_amdgcn_s_wait_asynccnt)
    __builtin_amdgcn_s_wait_asynccnt(0);
#else
    asm volatile("s_wait_asynccnt 0" ::: "memory");
#endif
}

__device__ __forceinline__ unsigned char f32_to_e4m3_sw(float f) {
    float c = fminf(fmaxf(f, -448.0f), 448.0f);
    unsigned u    = __float_as_uint(c);
    unsigned sign = (u >> 24) & 0x80u;
    unsigned au   = u & 0x7FFFFFFFu;
    if (au < 0x3C800000u) {
        float af = __uint_as_float(au);
        int m = __float2int_rn(af * 512.0f);
        return (unsigned char)(sign | (unsigned)m);
    }
    unsigned lsb = (au >> 20) & 1u;
    au += 0x0007FFFFu + lsb;
    unsigned exp = ((au >> 23) & 0xFFu) - 127u + 7u;
    unsigned man = (au >> 20) & 7u;
    return (unsigned char)(sign | (exp << 3) | man);
}

__device__ __forceinline__ unsigned pack4_e4m3(float x0, float x1, float x2, float x3) {
#if __has_builtin(__builtin_amdgcn_cvt_pk_fp8_f32)
    x0 = fminf(fmaxf(x0, -448.0f), 448.0f);
    x1 = fminf(fmaxf(x1, -448.0f), 448.0f);
    x2 = fminf(fmaxf(x2, -448.0f), 448.0f);
    x3 = fminf(fmaxf(x3, -448.0f), 448.0f);
    int p = 0;
    p = __builtin_amdgcn_cvt_pk_fp8_f32(x0, x1, p, false);
    p = __builtin_amdgcn_cvt_pk_fp8_f32(x2, x3, p, true);
    return (unsigned)p;
#else
    return (unsigned)f32_to_e4m3_sw(x0)
         | ((unsigned)f32_to_e4m3_sw(x1) << 8)
         | ((unsigned)f32_to_e4m3_sw(x2) << 16)
         | ((unsigned)f32_to_e4m3_sw(x3) << 24);
#endif
}

__global__ __launch_bounds__(256)
void quant_fp8_kernel(const float* __restrict__ src,
                      const float* __restrict__ scale,
                      unsigned char* __restrict__ dst,
                      long long n) {
    long long i = ((long long)blockIdx.x * blockDim.x + threadIdx.x) * 8;
    if (i >= n) return;
    float rs = (scale != nullptr) ? (1.0f / scale[0]) : 1.0f;
    const float4* s = (const float4*)(src + i);
    float4 a = s[0];
    float4 b = s[1];
    uint2 r;
    r.x = pack4_e4m3(a.x * rs, a.y * rs, a.z * rs, a.w * rs);
    r.y = pack4_e4m3(b.x * rs, b.y * rs, b.z * rs, b.w * rs);
    *(uint2*)(dst + i) = r;
}

__device__ __forceinline__ void compute_tile(const unsigned char* __restrict__ aTile,
                                             const unsigned char* __restrict__ bTile,
                                             int waveM, int waveN, int lrow, int halfK,
                                             v8f acc[2][4]) {
    v16i afrag[2];
#pragma unroll
    for (int m = 0; m < 2; ++m) {
        const unsigned char* ap = aTile + (waveM * 32 + m * 16 + lrow) * BK + (halfK << 4);
#pragma unroll
        for (int j = 0; j < 8; ++j) {
            unsigned long long d = *(const unsigned long long*)(ap + (j >> 1) * 32 + (j & 1) * 8);
            afrag[m][2 * j]     = (int)(unsigned)d;
            afrag[m][2 * j + 1] = (int)(unsigned)(d >> 32);
        }
    }
    v16i bfrag[4];
#pragma unroll
    for (int n = 0; n < 4; ++n) {
        const unsigned char* bp = bTile + (waveN * 64 + n * 16 + lrow) * BK + (halfK << 4);
#pragma unroll
        for (int j = 0; j < 4; ++j) {
            uint4 q = *(const uint4*)(bp + j * 32);
            bfrag[n][4 * j]     = (int)q.x;
            bfrag[n][4 * j + 1] = (int)q.y;
            bfrag[n][4 * j + 2] = (int)q.z;
            bfrag[n][4 * j + 3] = (int)q.w;
        }
    }
#define WMMA_MN(m_, n_, ra_)                                                  \
    acc[m_][n_] = __builtin_amdgcn_wmma_f32_16x16x128_fp8_fp8(                \
        afrag[m_], bfrag[n_], (short)0, acc[m_][n_], ra_, false)
    WMMA_MN(0, 0, false);
    WMMA_MN(0, 1, true);
    WMMA_MN(0, 2, true);
    WMMA_MN(0, 3, true);
    WMMA_MN(1, 0, false);
    WMMA_MN(1, 1, true);
    WMMA_MN(1, 2, true);
    WMMA_MN(1, 3, true);
#undef WMMA_MN
}

__global__ __launch_bounds__(GEMM_THREADS)
void fp8_gemm_kernel(const unsigned char* __restrict__ qx,
                     const unsigned char* __restrict__ qw,
                     const float* __restrict__ bias,
                     const float* __restrict__ input_scale,
                     const float* __restrict__ weight_scale,
                     float* __restrict__ out,
                     int M, int N, int K) {
    const int tid   = threadIdx.x;
    const int lane  = tid & 31;
    const int wave  = tid >> 5;
    const int waveM = wave >> 1;
    const int waveN = wave & 1;
    const int halfK = (lane >> 4);
    const int lrow  = lane & 15;

    const int blockRow = blockIdx.y * BM;
    const int blockCol = blockIdx.x * BN;
    const int nIter    = K / BK;

    v8f acc[2][4];
#pragma unroll
    for (int m = 0; m < 2; ++m)
#pragma unroll
        for (int n = 0; n < 4; ++n)
            acc[m][n] = (v8f)(0.0f);

#if USE_ASYNC_LDS
    __shared__ unsigned char ldsA[2][BM * BK];
    __shared__ unsigned char ldsB[2][BN * BK];

    auto stage = [&](int buf, int kb) {
#pragma unroll
        for (int i = 0; i < 4; ++i) {
            int c   = tid + GEMM_THREADS * i;
            int row = c >> 3;
            int col = (c & 7) << 4;
            __builtin_amdgcn_global_load_async_to_lds_b128(
                (gptr_b128)(qx + (size_t)(blockRow + row) * K + kb + col),
                (lptr_b128)(&ldsA[buf][row * BK + col]), 0, 0);
            __builtin_amdgcn_global_load_async_to_lds_b128(
                (gptr_b128)(qw + (size_t)(blockCol + row) * K + kb + col),
                (lptr_b128)(&ldsB[buf][row * BK + col]), 0, 0);
        }
    };

    stage(0, 0);
    for (int it = 0; it < nIter; ++it) {
        const int cur = it & 1;
        wait_asynccnt0();
        __syncthreads();
        if (it + 1 < nIter) stage(cur ^ 1, (it + 1) * BK);
        compute_tile(&ldsA[cur][0], &ldsB[cur][0], waveM, waveN, lrow, halfK, acc);
    }
#else
    __shared__ unsigned char ldsA[BM * BK];
    __shared__ unsigned char ldsB[BN * BK];

    uint4 pa[4], pb[4];
    auto preload = [&](int kb) {
#pragma unroll
        for (int i = 0; i < 4; ++i) {
            int c   = tid + GEMM_THREADS * i;
            int row = c >> 3;
            int col = (c & 7) << 4;
            pa[i] = *(const uint4*)(qx + (size_t)(blockRow + row) * K + kb + col);
            pb[i] = *(const uint4*)(qw + (size_t)(blockCol + row) * K + kb + col);
        }
    };

    preload(0);
    for (int it = 0; it < nIter; ++it) {
#pragma unroll
        for (int i = 0; i < 4; ++i) {
            int c   = tid + GEMM_THREADS * i;
            int row = c >> 3;
            int col = (c & 7) << 4;
            *(uint4*)(ldsA + row * BK + col) = pa[i];
            *(uint4*)(ldsB + row * BK + col) = pb[i];
        }
        __syncthreads();
        if (it + 1 < nIter) preload((it + 1) * BK);
        compute_tile(ldsA, ldsB, waveM, waveN, lrow, halfK, acc);
        __syncthreads();
    }
#endif

    const float scale = input_scale[0] * weight_scale[0];
#pragma unroll
    for (int m = 0; m < 2; ++m) {
        int row0 = blockRow + waveM * 32 + m * 16 + (halfK << 3);
#pragma unroll
        for (int n = 0; n < 4; ++n) {
            int col  = blockCol + waveN * 64 + n * 16 + lrow;
            float bv = bias[col];
            float* p = out + (size_t)row0 * N + col;
#pragma unroll
            for (int r = 0; r < 8; ++r)
                p[(size_t)r * N] = acc[m][n][r] * scale + bv;
        }
    }
}

extern "C" void kernel_launch(void* const* d_in, const int* in_sizes, int n_in,
                              void* d_out, int out_size, void* d_ws, size_t ws_size,
                              hipStream_t stream) {
    const float* x        = (const float*)d_in[0];
    const float* w        = (const float*)d_in[1];
    const float* w_scale  = (const float*)d_in[2];
    const float* bias     = (const float*)d_in[3];
    const float* in_scale = (const float*)d_in[4];
    float* out            = (float*)d_out;

    const long long N = in_sizes[3];
    const long long K = (long long)in_sizes[1] / N;
    const long long M = (long long)in_sizes[0] / K;

    unsigned char* qx = (unsigned char*)d_ws;
    unsigned char* qw = qx + (size_t)(M * K);

    const long long nx = M * K;
    const long long nw = N * K;
    quant_fp8_kernel<<<dim3((unsigned)((nx / 8 + 255) / 256)), 256, 0, stream>>>(
        x, in_scale, qx, nx);
    quant_fp8_kernel<<<dim3((unsigned)((nw / 8 + 255) / 256)), 256, 0, stream>>>(
        w, nullptr, qw, nw);

    dim3 grid((unsigned)(N / BN), (unsigned)(M / BM));
    fp8_gemm_kernel<<<grid, GEMM_THREADS, 0, stream>>>(
        qx, qw, bias, in_scale, w_scale, out, (int)M, (int)N, (int)K);
}
